// FaceDetection_20830591386199
// MI455X (gfx1250) — matrix operand order repaired and checked by reading, not compiled, not run on hardware
//
#include <hip/hip_runtime.h>
#include <hip/hip_bf16.h>

typedef __attribute__((ext_vector_type(16))) _Float16 v16h;
typedef __attribute__((ext_vector_type(8)))  _Float16 v8h;
typedef __attribute__((ext_vector_type(8)))  float    v8f;

#define TOPK   100
#define NANCH  12288
#define FHW    64

union V16 { v16h v; v8h h[2]; };

// ---------------- conv1: 3ch fp32 direct conv, stride 2, ReLU, NHWC f16 out ----------
// thread idx = p*32 + c : consecutive threads write consecutive channels (coalesced)
__global__ __launch_bounds__(256) void k_conv1(const float* __restrict__ frame,
                                               const float* __restrict__ w,
                                               const float* __restrict__ b,
                                               _Float16* __restrict__ out) {
  int idx = blockIdx.x * blockDim.x + threadIdx.x;   // 512*512*32 exactly
  int c = idx & 31;
  int p = idx >> 5;
  int oy = p >> 9, ox = p & 511;
  float acc = b[c];
  for (int ci = 0; ci < 3; ++ci)
    for (int fy = 0; fy < 3; ++fy) {
      int iy = oy * 2 + fy;                 // SAME, stride2: pad_lo = 0
      if (iy >= 1024) continue;
      for (int fx = 0; fx < 3; ++fx) {
        int ix = ox * 2 + fx;
        if (ix >= 1024) continue;
        acc += frame[ci * 1048576 + iy * 1024 + ix] * w[((c * 3 + ci) * 3 + fy) * 3 + fx];
      }
    }
  out[p * 32 + c] = (_Float16)fmaxf(acc, 0.0f);      // NHWC
}

// -------- weight packer: OIHW fp32 -> f16 [CoutPad][K], K index = (fy*3+fx)*Cin + ci --
__global__ void k_pack(const float* __restrict__ w, _Float16* __restrict__ wp,
                       int Cout, int CoutPad, int Cin) {
  int K = Cin * 9;
  int total = CoutPad * K;
  for (int idx = blockIdx.x * blockDim.x + threadIdx.x; idx < total;
       idx += gridDim.x * blockDim.x) {
    int n = idx / K;
    int k = idx - n * K;
    int tap = k / Cin;
    int ci  = k - tap * Cin;
    int fy = tap / 3, fx = tap - fy * 3;
    float v = (n < Cout) ? w[((n * Cin + ci) * 3 + fy) * 3 + fx] : 0.0f;
    wp[idx] = (_Float16)v;
  }
}

// ---------------- implicit-GEMM conv via v_wmma_f32_16x16x32_f16 (LDS-free) ----------
// NHWC f16 in/out. One wave = 16 output pixels x 16 output channels.
// A fragment: lane nl holds pixel pbase+nl; each K=32 block is one conv tap's 32
// contiguous channels -> two aligned b128 global loads (zero on OOB padding).
// B fragment: lane nl holds packed-weight row nbase+nl -> two aligned b128 loads.
__global__ __launch_bounds__(128) void k_conv_wmma(
    const _Float16* __restrict__ in, const _Float16* __restrict__ wp,
    const float* __restrict__ bias, _Float16* __restrict__ out,
    int Cin, int IH, int IW, int OH, int OW,
    int CoutPad, int CoutReal, int stride, int pad, int relu, int nPixTiles) {
  const int lane   = threadIdx.x & 31;
  const int wave   = threadIdx.x >> 5;
  const int laneHi = lane >> 4;            // 0: lanes 0-15, 1: lanes 16-31
  const int nl     = lane & 15;            // A-row (pixel) / B-row (channel) in tile
  const int nChunks = CoutPad >> 4;
  const int wid = blockIdx.x * 4 + wave;
  if (wid >= nPixTiles * nChunks) return;  // wave-uniform -> EXEC all-ones for WMMA
  const int ptile  = wid / nChunks;
  const int nchunk = wid - ptile * nChunks;
  const int pbase = ptile * 16;            // 16 pixels on one output row (OW%16==0)
  const int nbase = nchunk * 16;
  const int oy0 = pbase / OW;
  const int ox0 = pbase - oy0 * OW;
  const int ax  = (ox0 + nl) * stride - pad;   // input x for this lane's pixel
  const int ay0 = oy0 * stride - pad;
  const int K = Cin * 9;
  const int cSteps = Cin >> 5;             // Cin is a multiple of 32 on all layers
  const _Float16* brow = wp + (size_t)(nbase + nl) * K;
  const v8h zero = {};

  v8f acc = {};
  int kg = 0;
  for (int fy = 0; fy < 3; ++fy) {
    int iy = ay0 + fy;
    bool yok = (unsigned)iy < (unsigned)IH;
    for (int fx = 0; fx < 3; ++fx) {
      int ix = ax + fx;
      bool ok = yok && ((unsigned)ix < (unsigned)IW);
      const _Float16* arow = in + ((size_t)iy * IW + ix) * Cin;   // NHWC pixel base
      __builtin_prefetch(brow + kg + 128, 0, 1);                  // global_prefetch B
      for (int cb = 0; cb < cSteps; ++cb, kg += 32) {
        V16 a, b;
        if (ok) {
          a.h[0] = *(const v8h*)(arow + cb * 32 + laneHi * 8);       // K +0..7 / +8..15
          a.h[1] = *(const v8h*)(arow + cb * 32 + laneHi * 8 + 16);  // K +16..23 / +24..31
        } else {
          a.h[0] = zero;
          a.h[1] = zero;
        }
        b.h[0] = *(const v8h*)(brow + kg + laneHi * 8);
        b.h[1] = *(const v8h*)(brow + kg + laneHi * 8 + 16);
        acc = __builtin_amdgcn_wmma_f32_16x16x32_f16(
            false, a.v, false, b.v, (short)0, acc, false, false);
      }
    }
  }
  float bn = (nbase + nl < CoutReal) ? bias[nbase + nl] : 0.0f;
#pragma unroll
  for (int r = 0; r < 8; ++r) {
    int M = r + laneHi * 8;                // D layout: lanes16-31 -> M+8
    float v = acc[r] + bn;
    if (relu) v = fmaxf(v, 0.0f);
    out[(size_t)(pbase + M) * CoutPad + nbase + nl] = (_Float16)v;   // NHWC
  }
}

// ---------------- anchor decode + sigmoid (head is NHWC [4096][16]) ----------------
__global__ void k_decode(const _Float16* __restrict__ head, float* __restrict__ boxes,
                         float* __restrict__ scores) {
  int i = blockIdx.x * blockDim.x + threadIdx.x;
  if (i >= NANCH) return;
  int a = i % 3; int xy = i / 3;
  int x = xy % FHW; int y = xy / FHW;
  int off = y * FHW + x;
  const _Float16* hp = head + off * 16 + a * 5;    // 5 contiguous channels
  float t0 = (float)hp[0];
  float t1 = (float)hp[1];
  float t2 = (float)hp[2];
  float t3 = (float)hp[3];
  float lg = (float)hp[4];
  float s = (a == 0) ? 32.0f : ((a == 1) ? 64.0f : 128.0f);
  float cx = (x + 0.5f) * 16.0f + t0 * s;
  float cy = (y + 0.5f) * 16.0f + t1 * s;
  float bw = s * expf(fminf(fmaxf(t2, -4.0f), 4.0f));
  float bh = s * expf(fminf(fmaxf(t3, -4.0f), 4.0f));
  boxes[i * 4 + 0] = cx - bw * 0.5f;
  boxes[i * 4 + 1] = cy - bh * 0.5f;
  boxes[i * 4 + 2] = cx + bw * 0.5f;
  boxes[i * 4 + 3] = cy + bh * 0.5f;
  scores[i] = 1.0f / (1.0f + expf(-lg));
}

// ---------------- top-K=100 (deterministic, tie-break by lowest index) ----------------
__global__ __launch_bounds__(256) void k_topk(const float* __restrict__ scores,
                                              const float* __restrict__ boxes,
                                              float* __restrict__ topbox,
                                              int* __restrict__ valid) {
  __shared__ unsigned sel[NANCH / 32];
  __shared__ float rs[256];
  __shared__ int ri[256];
  int tid = threadIdx.x;
  for (int i = tid; i < NANCH / 32; i += 256) sel[i] = 0u;
  __syncthreads();
  for (int t = 0; t < TOPK; ++t) {
    float best = -2.0f; int bi = NANCH;
    for (int i = tid; i < NANCH; i += 256) {
      if (sel[i >> 5] & (1u << (i & 31))) continue;
      float sc = scores[i];
      float m = (sc >= 0.5f) ? sc : -1.0f;
      if (m > best || (m == best && i < bi)) { best = m; bi = i; }
    }
    rs[tid] = best; ri[tid] = bi;
    __syncthreads();
    for (int s2 = 128; s2 > 0; s2 >>= 1) {
      if (tid < s2) {
        if (rs[tid + s2] > rs[tid] ||
            (rs[tid + s2] == rs[tid] && ri[tid + s2] < ri[tid])) {
          rs[tid] = rs[tid + s2]; ri[tid] = ri[tid + s2];
        }
      }
      __syncthreads();
    }
    if (tid == 0) {
      int s3 = ri[0];
      int si = (s3 < NANCH) ? s3 : 0;
      sel[si >> 5] |= (1u << (si & 31));
      topbox[t * 4 + 0] = boxes[si * 4 + 0];
      topbox[t * 4 + 1] = boxes[si * 4 + 1];
      topbox[t * 4 + 2] = boxes[si * 4 + 2];
      topbox[t * 4 + 3] = boxes[si * 4 + 3];
      valid[t] = (rs[0] > 0.0f) ? 1 : 0;
    }
    __syncthreads();
  }
}

// ---------------- sequential NMS recurrence ----------------
__global__ __launch_bounds__(128) void k_nms(const float* __restrict__ topbox,
                                             const int* __restrict__ valid,
                                             int* __restrict__ keepOut) {
  __shared__ float bx[TOPK * 4];
  __shared__ float area[TOPK];
  __shared__ int keep[TOPK];
  int tid = threadIdx.x;
  for (int i = tid; i < TOPK * 4; i += 128) bx[i] = topbox[i];
  __syncthreads();
  if (tid < TOPK) {
    area[tid] = (bx[tid * 4 + 2] - bx[tid * 4 + 0]) * (bx[tid * 4 + 3] - bx[tid * 4 + 1]);
    keep[tid] = valid[tid];
  }
  __syncthreads();
  for (int i = 0; i < TOPK; ++i) {
    if (tid < TOPK && tid > i && keep[i]) {
      float lx = fmaxf(bx[i * 4 + 0], bx[tid * 4 + 0]);
      float ly = fmaxf(bx[i * 4 + 1], bx[tid * 4 + 1]);
      float rx = fminf(bx[i * 4 + 2], bx[tid * 4 + 2]);
      float ry = fminf(bx[i * 4 + 3], bx[tid * 4 + 3]);
      float iw = fmaxf(rx - lx, 0.0f), ih = fmaxf(ry - ly, 0.0f);
      float inter = iw * ih;
      float iou = inter / (area[i] + area[tid] - inter + 1e-9f);
      if (iou > 0.5f) keep[tid] = 0;
    }
    __syncthreads();
  }
  if (tid < TOPK) keepOut[tid] = keep[tid];
}

// ---------------- draw red borders onto the frame ----------------
__global__ __launch_bounds__(256) void k_draw(const float* __restrict__ frame,
                                              const float* __restrict__ topbox,
                                              const int* __restrict__ keep,
                                              float* __restrict__ out) {
  __shared__ float bx[TOPK * 4];
  __shared__ int kp[TOPK];
  int tid = threadIdx.x;
  for (int i = tid; i < TOPK; i += 256) {
    bx[i * 4 + 0] = fminf(fmaxf(topbox[i * 4 + 0], 0.0f), 1024.0f);
    bx[i * 4 + 1] = fminf(fmaxf(topbox[i * 4 + 1], 0.0f), 1024.0f);
    bx[i * 4 + 2] = fminf(fmaxf(topbox[i * 4 + 2], 0.0f), 1024.0f);
    bx[i * 4 + 3] = fminf(fmaxf(topbox[i * 4 + 3], 0.0f), 1024.0f);
    kp[i] = keep[i];
  }
  __syncthreads();
  int p = blockIdx.x * 256 + tid;
  float fy = (float)(p >> 10), fx = (float)(p & 1023);
  bool m = false;
  for (int j = 0; j < TOPK; ++j) {
    if (!kp[j]) continue;
    float x1 = bx[j * 4 + 0], y1 = bx[j * 4 + 1];
    float x2 = bx[j * 4 + 2], y2 = bx[j * 4 + 3];
    bool inside = (fy >= y1) && (fy < y2) && (fx >= x1) && (fx < x2);
    bool inner = (fy >= y1 + 4.0f) && (fy < y2 - 4.0f) &&
                 (fx >= x1 + 4.0f) && (fx < x2 - 4.0f);
    if (inside && !inner) { m = true; break; }
  }
  const int HW = 1024 * 1024;
  out[p]          = m ? 1.0f : frame[p];
  out[HW + p]     = m ? 0.0f : frame[HW + p];
  out[2 * HW + p] = m ? 0.0f : frame[2 * HW + p];
}

extern "C" void kernel_launch(void* const* d_in, const int* in_sizes, int n_in,
                              void* d_out, int out_size, void* d_ws, size_t ws_size,
                              hipStream_t stream) {
  const float* frame = (const float*)d_in[0];
  const float* w1 = (const float*)d_in[1];  const float* b1 = (const float*)d_in[2];
  const float* w2 = (const float*)d_in[3];  const float* b2 = (const float*)d_in[4];
  const float* w3 = (const float*)d_in[5];  const float* b3 = (const float*)d_in[6];
  const float* w4 = (const float*)d_in[7];  const float* b4 = (const float*)d_in[8];
  const float* wh = (const float*)d_in[9];  const float* bh = (const float*)d_in[10];
  float* out = (float*)d_out;

  char* ws = (char*)d_ws;
  size_t off = 0;
  auto bump = [&](size_t bytes) -> void* {
    void* p = (void*)(ws + off);
    off += (bytes + 255) & ~(size_t)255;
    return p;
  };
  _Float16* x1  = (_Float16*)bump(512ull * 512 * 32 * 2);   // NHWC
  _Float16* x2  = (_Float16*)bump(256ull * 256 * 64 * 2);
  _Float16* x3  = (_Float16*)bump(128ull * 128 * 128 * 2);
  _Float16* x4  = (_Float16*)bump(64ull * 64 * 256 * 2);
  _Float16* xh  = (_Float16*)bump(64ull * 64 * 16 * 2);
  _Float16* w2p = (_Float16*)bump(64ull * 288 * 2);
  _Float16* w3p = (_Float16*)bump(128ull * 576 * 2);
  _Float16* w4p = (_Float16*)bump(256ull * 1152 * 2);
  _Float16* whp = (_Float16*)bump(16ull * 2304 * 2);
  float* boxes  = (float*)bump((size_t)NANCH * 4 * 4);
  float* scores = (float*)bump((size_t)NANCH * 4);
  float* topbox = (float*)bump(TOPK * 4 * 4);
  int*   tvalid = (int*)bump(TOPK * 4);
  int*   keep   = (int*)bump(TOPK * 4);

  // weight packing (fp32 OIHW -> f16 [CoutPad][9*Cin], tap-major / channel-minor)
  k_pack<<<64, 256, 0, stream>>>(w2, w2p, 64, 64, 32);
  k_pack<<<128, 256, 0, stream>>>(w3, w3p, 128, 128, 64);
  k_pack<<<512, 256, 0, stream>>>(w4, w4p, 256, 256, 128);
  k_pack<<<64, 256, 0, stream>>>(wh, whp, 15, 16, 256);

  // conv1: 3 -> 32, 1024 -> 512 (direct fp32, NHWC f16 out)
  k_conv1<<<(512 * 512 * 32) / 256, 256, 0, stream>>>(frame, w1, b1, x1);

  // WMMA implicit-GEMM layers: grid = ceil(pixTiles*chunks / 4 waves per block)
  // conv2: 32 -> 64, 512 -> 256
  {
    int pt = 256 * 256 / 16, tw = pt * (64 / 16);
    k_conv_wmma<<<(tw + 3) / 4, 128, 0, stream>>>(
        x1, w2p, b2, x2, 32, 512, 512, 256, 256, 64, 64, 2, 0, 1, pt);
  }
  // conv3: 64 -> 128, 256 -> 128
  {
    int pt = 128 * 128 / 16, tw = pt * (128 / 16);
    k_conv_wmma<<<(tw + 3) / 4, 128, 0, stream>>>(
        x2, w3p, b3, x3, 64, 256, 256, 128, 128, 128, 128, 2, 0, 1, pt);
  }
  // conv4: 128 -> 256, 128 -> 64
  {
    int pt = 64 * 64 / 16, tw = pt * (256 / 16);
    k_conv_wmma<<<(tw + 3) / 4, 128, 0, stream>>>(
        x3, w4p, b4, x4, 128, 128, 128, 64, 64, 256, 256, 2, 0, 1, pt);
  }
  // head: 256 -> 15 (padded 16), stride 1, pad 1, no ReLU
  {
    int pt = 64 * 64 / 16, tw = pt * 1;
    k_conv_wmma<<<(tw + 3) / 4, 128, 0, stream>>>(
        x4, whp, bh, xh, 256, 64, 64, 64, 64, 16, 15, 1, 1, 0, pt);
  }

  k_decode<<<NANCH / 256, 256, 0, stream>>>(xh, boxes, scores);
  k_topk<<<1, 256, 0, stream>>>(scores, boxes, topbox, tvalid);
  k_nms<<<1, 128, 0, stream>>>(topbox, tvalid, keep);
  k_draw<<<(1024 * 1024) / 256, 256, 0, stream>>>(frame, topbox, keep, out);
}
